// LLaDA_51307679318302
// MI455X (gfx1250) — matrix operand order repaired and checked by reading, not compiled, not run on hardware
//
#include <hip/hip_runtime.h>

#define S 2048
#define DIM 1024
#define NH 16
#define HD 64
#define HID 2816
#define VOCAB 256
#define NL 4

typedef __attribute__((ext_vector_type(16))) __bf16 v16bf;
typedef __attribute__((ext_vector_type(8)))  __bf16 v8bf;
typedef __attribute__((ext_vector_type(8)))  float  v8f;

// ---------------- WMMA fragment helpers (contiguous k order, same for A and B) -------------
// A (16x32 bf16, M x K): lane m=lane&15 holds K = (lane>>4)*16 .. +16 contiguous.
__device__ __forceinline__ v16bf load_a_frag(const __bf16* p0, int lda) {
  int lane = threadIdx.x & 31;
  int m  = lane & 15;
  int ko = (lane >> 4) << 4;                 // 0 or 16
  const __bf16* p = p0 + m * lda + ko;
  v8bf c0 = *(const v8bf*)(p);
  v8bf c1 = *(const v8bf*)(p + 8);
  v16bf r;
#pragma unroll
  for (int i = 0; i < 8; ++i) { r[i] = c0[i]; r[i + 8] = c1[i]; }
  return r;
}

// B (32x16 bf16, K x N) supplied as N x K row-major storage (weight W[n][k]):
// lane n=lane&15 holds K = (lane>>4)*16 .. +16 contiguous.
__device__ __forceinline__ v16bf load_b_frag(const __bf16* p0, int ldb) {
  int lane = threadIdx.x & 31;
  int n  = lane & 15;
  int ko = (lane >> 4) << 4;                 // 0 or 16
  return *(const v16bf*)(p0 + n * ldb + ko);
}

__device__ __forceinline__ v8f wmma_bf16(v16bf a, v16bf b, v8f c) {
  return __builtin_amdgcn_wmma_f32_16x16x32_bf16(false, a, false, b, (short)0, c, false, false);
}

// ---------------- GEMM: C[M,N] (f32) = A[M,K](bf16) x W[N,K](bf16)^T (+ res) -------------
// 256 threads = 8 waves; wave -> 32x64 tile; K-loop ping-pong unrolled x2 (two disjoint
// fragment register sets, no rotation copies); block -> 64x256 tile.
__global__ __launch_bounds__(256) void k_gemm(const __bf16* __restrict__ A,
                                              const __bf16* __restrict__ W,
                                              float* C, const float* res,
                                              int M, int N, int K) {
  int wave  = threadIdx.x >> 5;
  int mBase = blockIdx.y * 64  + (wave & 1) * 32;
  int nBase = blockIdx.x * 256 + (wave >> 1) * 64;

  const __bf16* Ap0 = A + (size_t)mBase * K;
  const __bf16* Ap1 = A + (size_t)(mBase + 16) * K;
  const __bf16* Wp0 = W + (size_t)(nBase)      * K;
  const __bf16* Wp1 = W + (size_t)(nBase + 16) * K;
  const __bf16* Wp2 = W + (size_t)(nBase + 32) * K;
  const __bf16* Wp3 = W + (size_t)(nBase + 48) * K;

  v8f acc[8] = {};

  // fragment set A: K-step 0
  v16bf a0 = load_a_frag(Ap0, K);
  v16bf a1 = load_a_frag(Ap1, K);
  v16bf b0 = load_b_frag(Wp0, K);
  v16bf b1 = load_b_frag(Wp1, K);
  v16bf b2 = load_b_frag(Wp2, K);
  v16bf b3 = load_b_frag(Wp3, K);

  // K/32 is even for all shapes used (K in {1024, 2816})
  for (int k = 32; k < K; k += 64) {
    // fragment set B: K-step k
    v16bf e0 = load_a_frag(Ap0 + k, K);
    v16bf e1 = load_a_frag(Ap1 + k, K);
    v16bf f0 = load_b_frag(Wp0 + k, K);
    v16bf f1 = load_b_frag(Wp1 + k, K);
    v16bf f2 = load_b_frag(Wp2 + k, K);
    v16bf f3 = load_b_frag(Wp3 + k, K);
    if (k + 96 < K) {                         // prime L2/WGP$ ahead of the pipeline
      __builtin_prefetch(Wp0 + k + 96, 0, 1);
      __builtin_prefetch(Wp2 + k + 96, 0, 1);
      __builtin_prefetch(Ap0 + k + 96, 0, 1);
    }
    // consume set A (step k-32)
    acc[0] = wmma_bf16(a0, b0, acc[0]);
    acc[1] = wmma_bf16(a1, b0, acc[1]);
    acc[2] = wmma_bf16(a0, b1, acc[2]);
    acc[3] = wmma_bf16(a1, b1, acc[3]);
    acc[4] = wmma_bf16(a0, b2, acc[4]);
    acc[5] = wmma_bf16(a1, b2, acc[5]);
    acc[6] = wmma_bf16(a0, b3, acc[6]);
    acc[7] = wmma_bf16(a1, b3, acc[7]);
    // refill set A with K-step k+32 (skipped only on the final pair)
    if (k + 32 < K) {
      a0 = load_a_frag(Ap0 + k + 32, K);
      a1 = load_a_frag(Ap1 + k + 32, K);
      b0 = load_b_frag(Wp0 + k + 32, K);
      b1 = load_b_frag(Wp1 + k + 32, K);
      b2 = load_b_frag(Wp2 + k + 32, K);
      b3 = load_b_frag(Wp3 + k + 32, K);
    }
    // consume set B (step k)
    acc[0] = wmma_bf16(e0, f0, acc[0]);
    acc[1] = wmma_bf16(e1, f0, acc[1]);
    acc[2] = wmma_bf16(e0, f1, acc[2]);
    acc[3] = wmma_bf16(e1, f1, acc[3]);
    acc[4] = wmma_bf16(e0, f2, acc[4]);
    acc[5] = wmma_bf16(e1, f2, acc[5]);
    acc[6] = wmma_bf16(e0, f3, acc[6]);
    acc[7] = wmma_bf16(e1, f3, acc[7]);
  }

  int lane = threadIdx.x & 31;
  int col = lane & 15, r0 = (lane >> 4) * 8;
#pragma unroll
  for (int j = 0; j < 8; ++j) {
    int mrow = mBase + (j & 1) * 16 + r0;
    int ncol = nBase + (j >> 1) * 16 + col;
#pragma unroll
    for (int r = 0; r < 8; ++r) {
      size_t idx = (size_t)(mrow + r) * N + ncol;
      float v = acc[j][r];
      if (res) v += res[idx];
      C[idx] = v;
    }
  }
}

// ---------------- Flash attention: one wave per (head, 16-query tile) -------------
// Q,Kc: [S][NH*HD] bf16 (RoPE applied). Vt: [NH*HD][S] bf16 (pre-transposed). O: [S][NH*HD] bf16.
__device__ __forceinline__ void attn_step(int kb, int hoff, int col, int hf,
                                          const __bf16* __restrict__ Vt, __bf16* P,
                                          v16bf qa0, v16bf qa1,
                                          v16bf kf0, v16bf kf1, v16bf kf2, v16bf kf3,
                                          v8f* o, float* mrow, float* lrow) {
  const float scale = 0.125f;                       // 1/sqrt(64)
  // V tiles (consumed after softmax -> latency hidden by exp/shuffle VALU work)
  v16bf vb0 = load_b_frag(Vt + (size_t)(hoff)      * S + kb, S);
  v16bf vb1 = load_b_frag(Vt + (size_t)(hoff + 16) * S + kb, S);
  v16bf vb2 = load_b_frag(Vt + (size_t)(hoff + 32) * S + kb, S);
  v16bf vb3 = load_b_frag(Vt + (size_t)(hoff + 48) * S + kb, S);

  v8f s0 = {}, s1 = {};
  s0 = wmma_bf16(qa0, kf0, s0);
  s0 = wmma_bf16(qa1, kf1, s0);
  s1 = wmma_bf16(qa0, kf2, s1);
  s1 = wmma_bf16(qa1, kf3, s1);

  float alpha[8];
#pragma unroll
  for (int r = 0; r < 8; ++r) {
    float a0 = s0[r] * scale, a1 = s1[r] * scale;
    float mx = fmaxf(a0, a1);
#pragma unroll
    for (int xm = 8; xm >= 1; xm >>= 1) mx = fmaxf(mx, __shfl_xor(mx, xm, 32));
    float mn = fmaxf(mrow[r], mx);
    float al = __expf(mrow[r] - mn);
    float p0 = __expf(a0 - mn), p1 = __expf(a1 - mn);
    float ps = p0 + p1;
#pragma unroll
    for (int xm = 8; xm >= 1; xm >>= 1) ps += __shfl_xor(ps, xm, 32);
    lrow[r] = lrow[r] * al + ps;
    mrow[r] = mn;
    alpha[r] = al;
    P[(hf * 8 + r) * 32 + col]      = (__bf16)p0;   // C-layout -> row-major LDS
    P[(hf * 8 + r) * 32 + col + 16] = (__bf16)p1;
  }
  __syncthreads();                                  // stores visible before A-layout reload
  v16bf pa = load_a_frag(P, 32);
  __syncthreads();                                  // loads done before next step's stores

#pragma unroll
  for (int r = 0; r < 8; ++r) {
    o[0][r] *= alpha[r]; o[1][r] *= alpha[r];
    o[2][r] *= alpha[r]; o[3][r] *= alpha[r];
  }
  o[0] = wmma_bf16(pa, vb0, o[0]);
  o[1] = wmma_bf16(pa, vb1, o[1]);
  o[2] = wmma_bf16(pa, vb2, o[2]);
  o[3] = wmma_bf16(pa, vb3, o[3]);
}

__global__ __launch_bounds__(32) void k_attn(const __bf16* __restrict__ Q,
                                             const __bf16* __restrict__ Kc,
                                             const __bf16* __restrict__ Vt,
                                             __bf16* __restrict__ O) {
  int hidx  = blockIdx.y;
  int hoff  = hidx * HD;
  int qBase = blockIdx.x * 16;
  __shared__ __bf16 P[16 * 32];
  int lane = threadIdx.x;
  int col = lane & 15, hf = lane >> 4;

  v16bf qa0 = load_a_frag(Q + (size_t)qBase * DIM + hoff,      DIM);
  v16bf qa1 = load_a_frag(Q + (size_t)qBase * DIM + hoff + 32, DIM);

  float mrow[8], lrow[8];
  v8f o[4] = {};
#pragma unroll
  for (int r = 0; r < 8; ++r) { mrow[r] = -3.0e38f; lrow[r] = 0.f; }

  // K-fragment set A: k-tile 0 (keys 0..31)
  v16bf kA0 = load_b_frag(Kc + hoff,                          DIM);
  v16bf kA1 = load_b_frag(Kc + hoff + 32,                     DIM);
  v16bf kA2 = load_b_frag(Kc + (size_t)16 * DIM + hoff,       DIM);
  v16bf kA3 = load_b_frag(Kc + (size_t)16 * DIM + hoff + 32,  DIM);

  // S/32 = 64 k-tiles: ping-pong unrolled x2
  for (int kb = 0; kb < S; kb += 64) {
    // set B: keys kb+32 .. kb+63
    v16bf kB0 = load_b_frag(Kc + (size_t)(kb + 32) * DIM + hoff,      DIM);
    v16bf kB1 = load_b_frag(Kc + (size_t)(kb + 32) * DIM + hoff + 32, DIM);
    v16bf kB2 = load_b_frag(Kc + (size_t)(kb + 48) * DIM + hoff,      DIM);
    v16bf kB3 = load_b_frag(Kc + (size_t)(kb + 48) * DIM + hoff + 32, DIM);

    attn_step(kb, hoff, col, hf, Vt, P, qa0, qa1, kA0, kA1, kA2, kA3, o, mrow, lrow);

    if (kb + 64 < S) {                        // refill set A with keys kb+64 .. kb+95
      kA0 = load_b_frag(Kc + (size_t)(kb + 64) * DIM + hoff,      DIM);
      kA1 = load_b_frag(Kc + (size_t)(kb + 64) * DIM + hoff + 32, DIM);
      kA2 = load_b_frag(Kc + (size_t)(kb + 80) * DIM + hoff,      DIM);
      kA3 = load_b_frag(Kc + (size_t)(kb + 80) * DIM + hoff + 32, DIM);
    }

    attn_step(kb + 32, hoff, col, hf, Vt, P, qa0, qa1, kB0, kB1, kB2, kB3, o, mrow, lrow);
  }

#pragma unroll
  for (int j = 0; j < 4; ++j)
#pragma unroll
    for (int r = 0; r < 8; ++r) {
      size_t row = qBase + hf * 8 + r;
      O[row * DIM + hoff + j * 16 + col] = (__bf16)(o[j][r] / lrow[r]);
    }
}

// ---------------- Elementwise / small kernels -------------
__global__ void k_embed(const int* __restrict__ tok, const float* __restrict__ emb,
                        float* __restrict__ h) {
  int s = blockIdx.x;
  int t = tok[s];
  for (int d = threadIdx.x; d < DIM; d += 256)
    h[(size_t)s * DIM + d] = emb[(size_t)t * DIM + d];
}

__global__ void k_rope_tables(float* cs, float* sn) {
  int i = blockIdx.x * 256 + threadIdx.x;
  if (i >= S * 32) return;
  int s = i >> 5, j = i & 31;
  float f = (float)s * powf(10000.f, -(float)j / 32.f);
  cs[i] = cosf(f);
  sn[i] = sinf(f);
}

__global__ void k_rope(const float* __restrict__ in, __bf16* __restrict__ out,
                       const float* __restrict__ cs, const float* __restrict__ sn) {
  int i = blockIdx.x * 256 + threadIdx.x;        // over S*NH*32
  int j = i & 31; int t = i >> 5; int hh = t & 15; int s = t >> 4;
  float c = cs[s * 32 + j], si = sn[s * 32 + j];
  size_t base = (size_t)s * DIM + hh * HD + 2 * j;
  float xr = in[base], xi = in[base + 1];
  out[base]     = (__bf16)(xr * c - xi * si);
  out[base + 1] = (__bf16)(xr * si + xi * c);
}

__global__ void k_vt(const float* __restrict__ v, __bf16* __restrict__ vt) {
  int i = blockIdx.x * 256 + threadIdx.x;        // over S*DIM
  int s = i >> 10; int d = i & 1023;
  vt[(size_t)d * S + s] = (__bf16)v[i];
}

__global__ void k_f2bf(const float* __restrict__ in, __bf16* __restrict__ out, int n) {
  int i = blockIdx.x * 256 + threadIdx.x;
  if (i < n) out[i] = (__bf16)in[i];
}

__global__ void k_silu_mul(const float* __restrict__ g, const float* __restrict__ u,
                           __bf16* __restrict__ y, int n) {
  int i = blockIdx.x * 256 + threadIdx.x;
  if (i < n) {
    float gg = g[i];
    float sg = gg / (1.f + __expf(-gg));
    y[i] = (__bf16)(sg * u[i]);
  }
}

__global__ __launch_bounds__(256) void k_rmsnorm(const float* __restrict__ x,
                                                 const float* __restrict__ w,
                                                 __bf16* __restrict__ out) {
  int row = blockIdx.x;
  const float* xr = x + (size_t)row * DIM;
  float ss = 0.f;
  for (int d = threadIdx.x; d < DIM; d += 256) { float v = xr[d]; ss += v * v; }
#pragma unroll
  for (int xm = 16; xm >= 1; xm >>= 1) ss += __shfl_xor(ss, xm, 32);
  __shared__ float red[8];
  if ((threadIdx.x & 31) == 0) red[threadIdx.x >> 5] = ss;
  __syncthreads();
  float tot = 0.f;
#pragma unroll
  for (int i = 0; i < 8; ++i) tot += red[i];
  float rs = rsqrtf(tot / DIM + 1e-5f);
  for (int d = threadIdx.x; d < DIM; d += 256)
    out[(size_t)row * DIM + d] = (__bf16)(xr[d] * rs * w[d]);
}

__global__ __launch_bounds__(256) void k_final(const float* __restrict__ h,
                                               const float* __restrict__ fw,
                                               const float* __restrict__ ow,
                                               float* __restrict__ out) {
  const float* x = h + (size_t)(S - 1) * DIM;
  int v = blockIdx.x;
  float ss = 0.f, dot = 0.f;
  for (int d = threadIdx.x; d < DIM; d += 256) {
    float xv = x[d];
    ss  += xv * xv;
    dot += xv * fw[d] * ow[(size_t)v * DIM + d];
  }
#pragma unroll
  for (int xm = 16; xm >= 1; xm >>= 1) {
    ss  += __shfl_xor(ss,  xm, 32);
    dot += __shfl_xor(dot, xm, 32);
  }
  __shared__ float r1[8], r2[8];
  if ((threadIdx.x & 31) == 0) { r1[threadIdx.x >> 5] = ss; r2[threadIdx.x >> 5] = dot; }
  __syncthreads();
  if (threadIdx.x == 0) {
    float t1 = 0.f, t2 = 0.f;
#pragma unroll
    for (int i = 0; i < 8; ++i) { t1 += r1[i]; t2 += r2[i]; }
    out[v] = t2 * rsqrtf(t1 / DIM + 1e-5f);
  }
}

// ---------------- Host orchestration -------------
extern "C" void kernel_launch(void* const* d_in, const int* in_sizes, int n_in,
                              void* d_out, int out_size, void* d_ws, size_t ws_size,
                              hipStream_t stream) {
  (void)in_sizes; (void)n_in; (void)out_size; (void)ws_size;
  const int*   tokens = (const int*)  d_in[0];
  const float* temb   = (const float*)d_in[1];
  const float* wq     = (const float*)d_in[2];
  const float* wk     = (const float*)d_in[3];
  const float* wv     = (const float*)d_in[4];
  const float* wo     = (const float*)d_in[5];
  const float* w1     = (const float*)d_in[6];
  const float* w2     = (const float*)d_in[7];
  const float* w3     = (const float*)d_in[8];
  const float* anw    = (const float*)d_in[9];
  const float* fnw    = (const float*)d_in[10];
  const float* finw   = (const float*)d_in[11];
  const float* ow     = (const float*)d_in[12];
  float* out = (float*)d_out;

  char* ws = (char*)d_ws;
  size_t off = 0;
  auto alloc = [&](size_t bytes) -> void* {
    off = (off + 255) & ~(size_t)255;
    void* p = ws + off;
    off += bytes;
    return p;
  };

  float*  hbuf = (float*) alloc((size_t)S * DIM * 4);
  __bf16* xn   = (__bf16*)alloc((size_t)S * DIM * 2);
  float*  qf   = (float*) alloc((size_t)S * DIM * 4);
  float*  kf   = (float*) alloc((size_t)S * DIM * 4);
  float*  vf   = (float*) alloc((size_t)S * DIM * 4);
  __bf16* qb   = (__bf16*)alloc((size_t)S * DIM * 2);
  __bf16* kb   = (__bf16*)alloc((size_t)S * DIM * 2);
  __bf16* vt   = (__bf16*)alloc((size_t)S * DIM * 2);
  __bf16* ab   = (__bf16*)alloc((size_t)S * DIM * 2);
  float*  g    = (float*) alloc((size_t)S * HID * 4);
  float*  u    = (float*) alloc((size_t)S * HID * 4);
  __bf16* yb   = (__bf16*)alloc((size_t)S * HID * 2);
  float*  cs   = (float*) alloc((size_t)S * 32 * 4);
  float*  sn   = (float*) alloc((size_t)S * 32 * 4);
  __bf16* wqb  = (__bf16*)alloc((size_t)DIM * DIM * 2);
  __bf16* wkb  = (__bf16*)alloc((size_t)DIM * DIM * 2);
  __bf16* wvb  = (__bf16*)alloc((size_t)DIM * DIM * 2);
  __bf16* wob  = (__bf16*)alloc((size_t)DIM * DIM * 2);
  __bf16* w1b  = (__bf16*)alloc((size_t)HID * DIM * 2);
  __bf16* w3b  = (__bf16*)alloc((size_t)HID * DIM * 2);
  __bf16* w2b  = (__bf16*)alloc((size_t)DIM * HID * 2);

  auto conv = [&](const float* src, __bf16* dst, int n) {
    k_f2bf<<<(n + 255) / 256, 256, 0, stream>>>(src, dst, n);
  };
  auto gemm = [&](const __bf16* A, const __bf16* W, float* C, const float* res,
                  int M, int N, int K) {
    dim3 gr(N / 256, M / 64);
    k_gemm<<<gr, 256, 0, stream>>>(A, W, C, res, M, N, K);
  };

  k_rope_tables<<<(S * 32 + 255) / 256, 256, 0, stream>>>(cs, sn);
  k_embed<<<S, 256, 0, stream>>>(tokens, temb, hbuf);

  for (int l = 0; l < NL; ++l) {
    conv(wq + (size_t)l * DIM * DIM, wqb, DIM * DIM);
    conv(wk + (size_t)l * DIM * DIM, wkb, DIM * DIM);
    conv(wv + (size_t)l * DIM * DIM, wvb, DIM * DIM);
    conv(wo + (size_t)l * DIM * DIM, wob, DIM * DIM);
    conv(w1 + (size_t)l * HID * DIM, w1b, HID * DIM);
    conv(w3 + (size_t)l * HID * DIM, w3b, HID * DIM);
    conv(w2 + (size_t)l * DIM * HID, w2b, DIM * HID);

    k_rmsnorm<<<S, 256, 0, stream>>>(hbuf, anw + (size_t)l * DIM, xn);
    gemm(xn, wqb, qf, nullptr, S, DIM, DIM);
    gemm(xn, wkb, kf, nullptr, S, DIM, DIM);
    gemm(xn, wvb, vf, nullptr, S, DIM, DIM);
    k_rope<<<(S * NH * 32) / 256, 256, 0, stream>>>(qf, qb, cs, sn);
    k_rope<<<(S * NH * 32) / 256, 256, 0, stream>>>(kf, kb, cs, sn);
    k_vt<<<(S * DIM) / 256, 256, 0, stream>>>(vf, vt);
    k_attn<<<dim3(S / 16, NH), 32, 0, stream>>>(qb, kb, vt, ab);
    gemm(ab, wob, hbuf, hbuf, S, DIM, DIM);

    k_rmsnorm<<<S, 256, 0, stream>>>(hbuf, fnw + (size_t)l * DIM, xn);
    gemm(xn, w1b, g, nullptr, S, HID, DIM);
    gemm(xn, w3b, u, nullptr, S, HID, DIM);
    k_silu_mul<<<(S * HID + 255) / 256, 256, 0, stream>>>(g, u, yb, S * HID);
    gemm(yb, w2b, hbuf, hbuf, S, DIM, HID);
  }

  k_final<<<VOCAB, 256, 0, stream>>>(hbuf, finw, ow, out);
}
